// CrfDecoder_71717363908808
// MI455X (gfx1250) — matrix operand order repaired and checked by reading, not compiled, not run on hardware
//
#include <hip/hip_runtime.h>
#include <hip/hip_bf16.h>

// ---------------------------------------------------------------------------
// CRF loss (logZ - gold) for B=64, S=1024, C=1, N=256 on gfx1250 (MI455X).
//
// Strategy: convert the per-step logsumexp recurrence into a linear-space
// GEMM  expA[16x256] * expT[256x256]  per 16-batch group, executed with
// v_wmma_f32_16x16x32_f16.  expT (f16, 128KB) lives in LDS pre-swizzled into
// B-fragment layout (only possible with CDNA5's 320KB/WGP LDS).  Alpha state
// and A fragments stay LDS-resident across all 1024 steps; emissions are
// streamed from HBM with software prefetch.
// ---------------------------------------------------------------------------

#define CRF_B   64
#define CRF_S   1024
#define CRF_N   256
#define CRF_BG  16          // batches per workgroup (== WMMA M)
#define CRF_NWG (CRF_B / CRF_BG)

typedef __attribute__((ext_vector_type(16))) _Float16 v16h;
typedef __attribute__((ext_vector_type(8)))  _Float16 v8h;
typedef __attribute__((ext_vector_type(8)))  float    v8f;

static __device__ __forceinline__ v16h cat16(v8h a, v8h b) {
    return __builtin_shufflevector(a, b, 0,1,2,3,4,5,6,7,8,9,10,11,12,13,14,15);
}

// LDS layout (bytes) for the forward kernel
#define OFF_T     0                         // expT swizzled: 16nt * 8kc * 32lane * 16h f16
#define SZ_T      (16*8*32*16*2)            // 131072
#define OFF_ALPHA (OFF_T + SZ_T)            // alpha f32 [16][256]
#define SZ_ALPHA  (16*256*4)                // 16384
#define OFF_A     (OFF_ALPHA + SZ_ALPHA)    // expA fragments f16: 8kc * 32lane * 16h
#define SZ_A      (8*32*16*2)               // 8192
#define OFF_M     (OFF_A + SZ_A)            // per-batch max, f32[16]
#define OFF_LEN   (OFF_M + 64)              // lengths, int[16]
#define SMEM_BYTES (OFF_LEN + 64)           // 155,776 B < 320KB/WGP

// ---------------------------------------------------------------------------
// Kernel 0: expT[k*N+n] = exp(transitions[k][n]) as f16 into workspace.
// ---------------------------------------------------------------------------
__global__ void crf_build_expT(const float* __restrict__ trans,
                               _Float16* __restrict__ expT) {
    int i = blockIdx.x * blockDim.x + threadIdx.x;
    if (i < CRF_N * CRF_N) expT[i] = (_Float16)__expf(trans[i]);
}

// ---------------------------------------------------------------------------
// Kernel 1: gold path score per batch (emit + trans + head + last).
// ---------------------------------------------------------------------------
__global__ void crf_gold(const float* __restrict__ emis,
                         const int*   __restrict__ targets,
                         const int*   __restrict__ lengths,
                         const float* __restrict__ trans,
                         const float* __restrict__ head,
                         const float* __restrict__ last,
                         float* __restrict__ gold) {
    __shared__ float red[256];
    const int b   = blockIdx.x;
    const int tid = threadIdx.x;
    const int len = lengths[b];
    float acc = 0.f;
    for (int t = tid; t < CRF_S; t += 256) {
        int tg = targets[b * CRF_S + t];
        if (t < len) acc += emis[(b * CRF_S + t) * CRF_N + tg];
        if (t >= 1 && t < len) {
            int tp = targets[b * CRF_S + t - 1];
            acc += trans[tp * CRF_N + tg];
        }
    }
    red[tid] = acc;
    __syncthreads();
    for (int o = 128; o > 0; o >>= 1) {
        if (tid < o) red[tid] += red[tid + o];
        __syncthreads();
    }
    if (tid == 0) {
        float g = red[0];
        g += head[targets[b * CRF_S]];
        g += last[targets[b * CRF_S + len - 1]];
        gold[b] = g;
    }
}

// ---------------------------------------------------------------------------
// Kernel 2: forward recurrence via WMMA + final logZ - gold.
// grid = 4 blocks (16 batches each), 256 threads = 8 waves.
// ---------------------------------------------------------------------------
__global__ __launch_bounds__(256, 1)
void crf_forward(const float*    __restrict__ emis,
                 const int*      __restrict__ lengths,
                 const _Float16* __restrict__ expT,
                 const float*    __restrict__ head,
                 const float*    __restrict__ last,
                 const float*    __restrict__ gold,
                 float*          __restrict__ out) {
    extern __shared__ char smem[];
    _Float16* sT     = (_Float16*)(smem + OFF_T);
    float*    sAlpha = (float*)   (smem + OFF_ALPHA);
    _Float16* sA     = (_Float16*)(smem + OFF_A);
    float*    sM     = (float*)   (smem + OFF_M);
    int*      sLen   = (int*)     (smem + OFF_LEN);

    const int tid  = threadIdx.x;
    const int bg   = blockIdx.x * CRF_BG;        // global batch base
    const int lane = tid & 31;
    const int wave = tid >> 5;                   // 0..7
    const int nt0  = wave * 2;                   // this wave's two N-tiles

    // ---- one-time: swizzle expT into B-fragment order in LDS -------------
    // B frag (32x16 f16): lane L -> N = nt*16 + (L&15); half h -> K = kc*32 + 8*(L>>4) + 16*(h>>3) + (h&7)
    for (int idx = tid; idx < CRF_N * CRF_N; idx += 256) {
        int h  = idx & 15;
        int ln = (idx >> 4) & 31;
        int kc = (idx >> 9) & 7;
        int nt = idx >> 12;
        int K  = kc * 32 + ((ln >> 4) << 3) + ((h >> 3) << 4) + (h & 7);
        int Nn = nt * 16 + (ln & 15);
        sT[idx] = expT[K * CRF_N + Nn];
    }
    if (tid < CRF_BG) sLen[tid] = lengths[bg + tid];

    // ---- alpha_0 = head + emissions[:,0] ---------------------------------
    for (int idx = tid; idx < CRF_BG * CRF_N; idx += 256) {
        int b = idx >> 8, j = idx & 255;
        sAlpha[idx] = head[j] + emis[(bg + b) * CRF_S * CRF_N + j];
    }
    __syncthreads();

    #pragma unroll 1
    for (int t = 1; t < CRF_S; ++t) {
        // (a) per-batch running max m_b over alpha[b,:]  (16 lanes per batch)
        {
            int b = tid >> 4, l = tid & 15;
            const float* row = sAlpha + b * CRF_N;
            float mx = -3.402823466e38f;
            #pragma unroll
            for (int j = 0; j < 16; ++j) mx = fmaxf(mx, row[l + j * 16]);
            for (int o = 8; o > 0; o >>= 1) mx = fmaxf(mx, __shfl_xor(mx, o, 16));
            if (l == 0) sM[b] = mx;
        }
        __syncthreads();

        // (b) expA = exp(alpha - m_b) written in A-fragment order
        // A frag (16x32 f16): lane L -> M = L&15; half h -> K = kc*32 + 8*(L>>4) + 16*(h>>3) + (h&7)
        {
            int kc = tid >> 5, L = tid & 31;
            int M  = L & 15;
            float mb = sM[M];
            const float* row = sAlpha + M * CRF_N;
            _Float16* dst = sA + ((kc * 32 + L) << 4);
            int kbase = kc * 32 + ((L >> 4) << 3);
            #pragma unroll
            for (int h = 0; h < 16; ++h) {
                int K = kbase + ((h >> 3) << 4) + (h & 7);
                dst[h] = (_Float16)__expf(row[K] - mb);
            }
        }
        __syncthreads();

        // (c) GEMM step: D[16 x 32] per wave = expA (16x256) * expT (256x32)
        v8f acc0 = {}; v8f acc1 = {};
        #pragma unroll
        for (int kc = 0; kc < 8; ++kc) {
            const v8h* ap = (const v8h*)(sA + ((kc * 32 + lane) << 4));
            v16h af = cat16(ap[0], ap[1]);
            const v8h* b0 = (const v8h*)(sT + (((nt0 * 8 + kc) * 32 + lane) << 4));
            v16h bf0 = cat16(b0[0], b0[1]);
            const v8h* b1 = (const v8h*)(sT + ((((nt0 + 1) * 8 + kc) * 32 + lane) << 4));
            v16h bf1 = cat16(b1[0], b1[1]);
            acc0 = __builtin_amdgcn_wmma_f32_16x16x32_f16(
                false, af, false, bf0, (short)0, acc0, false, false);
            acc1 = __builtin_amdgcn_wmma_f32_16x16x32_f16(
                false, af, false, bf1, (short)0, acc1, false, false);
        }

        // (d) prefetch next step's emissions (16 batches x 64B per lane group)
        if (t + 1 < CRF_S) {
            int pb = tid >> 4, pj = (tid & 15) << 4;
            __builtin_prefetch(
                &emis[((bg + pb) * CRF_S + (t + 1)) * CRF_N + pj], 0, 0);
        }

        // (e) epilogue: alpha_new = log(acc) + m_b + e_t ; masked by t < len
        // C/D layout: lane L -> j = tile*16 + (L&15); VGPR r -> b = 8*(L>>4) + r
        {
            int jc = lane & 15;
            int rb = (lane >> 4) << 3;
            int j0 = nt0 * 16 + jc;
            #pragma unroll
            for (int r = 0; r < 8; ++r) {
                int b  = rb + r;
                float mb = sM[b];
                const float* ep = &emis[((bg + b) * CRF_S + t) * CRF_N + j0];
                float na0 = __logf(acc0[r]) + mb + ep[0];
                float na1 = __logf(acc1[r]) + mb + ep[16];
                if (t < sLen[b]) {
                    sAlpha[b * CRF_N + j0]      = na0;
                    sAlpha[b * CRF_N + j0 + 16] = na1;
                }
            }
        }
        __syncthreads();
    }

    // ---- logZ = logsumexp(alpha + last);  out = logZ - gold --------------
    {
        int b = tid >> 4, l = tid & 15;
        const float* row = sAlpha + b * CRF_N;
        float mx = -3.402823466e38f;
        #pragma unroll
        for (int j = 0; j < 16; ++j) {
            int c = l + j * 16;
            mx = fmaxf(mx, row[c] + last[c]);
        }
        for (int o = 8; o > 0; o >>= 1) mx = fmaxf(mx, __shfl_xor(mx, o, 16));
        float s = 0.f;
        #pragma unroll
        for (int j = 0; j < 16; ++j) {
            int c = l + j * 16;
            s += __expf(row[c] + last[c] - mx);
        }
        for (int o = 8; o > 0; o >>= 1) s += __shfl_xor(s, o, 16);
        if (l == 0) out[bg + b] = (mx + __logf(s)) - gold[bg + b];
    }
}

// ---------------------------------------------------------------------------
// Host launcher.  Inputs (setup_inputs order):
//   0 emissions f32 [B,S,1,N], 1 targets i32 [B,S], 2 lengths i32 [B],
//   3 transitions f32 [1,N,N], 4 head f32 [1,N], 5 last f32 [1,N]
// Output: f32 [B,1].
// ---------------------------------------------------------------------------
extern "C" void kernel_launch(void* const* d_in, const int* in_sizes, int n_in,
                              void* d_out, int out_size, void* d_ws, size_t ws_size,
                              hipStream_t stream) {
    const float* emissions   = (const float*)d_in[0];
    const int*   targets     = (const int*)  d_in[1];
    const int*   lengths     = (const int*)  d_in[2];
    const float* transitions = (const float*)d_in[3];
    const float* head        = (const float*)d_in[4];
    const float* last        = (const float*)d_in[5];
    float*       out         = (float*)d_out;

    _Float16* expT = (_Float16*)d_ws;                               // 128 KB
    float*    gold = (float*)((char*)d_ws + CRF_N * CRF_N * 2);     // 64 floats

    crf_build_expT<<<(CRF_N * CRF_N + 255) / 256, 256, 0, stream>>>(
        transitions, expT);
    crf_gold<<<CRF_B, 256, 0, stream>>>(
        emissions, targets, lengths, transitions, head, last, gold);
    crf_forward<<<CRF_NWG, 256, SMEM_BYTES, stream>>>(
        emissions, lengths, expT, head, last, gold, out);
}
